// InContextGRU_10943576670284
// MI455X (gfx1250) — matrix operand order repaired and checked by reading, not compiled, not run on hardware
//
#include <hip/hip_runtime.h>
#include <hip/hip_bf16.h>

// ---------------------------------------------------------------------------
// Types / helpers
// ---------------------------------------------------------------------------
typedef __attribute__((ext_vector_type(16))) __bf16 v16bf;
typedef __attribute__((ext_vector_type(8)))  float  v8f;
typedef __attribute__((ext_vector_type(4)))  int    v4i;

union FragU { v16bf v; uint4 q[2]; __bf16 e[16]; };

// Hardware f32 -> bf16 convert (RNE); avoids the 4-op software rounding seq.
__device__ __forceinline__ unsigned short f2bf(float f) {
  __bf16 h = (__bf16)f;
  return __builtin_bit_cast(unsigned short, h);
}
__device__ __forceinline__ float bf2f(unsigned short h) {
  union { float f; unsigned u; } v; v.u = ((unsigned)h) << 16;
  return v.f;
}
__device__ __forceinline__ float sigmoidf(float x) {
  return 1.0f / (1.0f + __expf(-x));
}

__device__ __forceinline__ v8f wmma_bf16(v16bf a, v16bf b, v8f c) {
  // D = A(16x32 bf16) * B(32x16 bf16) + C(16x16 f32)
  return __builtin_amdgcn_wmma_f32_16x16x32_bf16(
      /*neg_a=*/false, a, /*neg_b=*/false, b,
      /*c_mod=*/(short)0, c, /*reuse_a=*/false, /*reuse_b=*/false);
}

// A fragment (16x32, bf16) from row-major [M][K] bf16 matrix.
__device__ __forceinline__ v16bf load_afrag(const unsigned short* A, int m0,
                                            int K, int k0, int lane) {
  int row = m0 + (lane & 15);
  int kk  = k0 + ((lane & 16) ? 16 : 0);
  const unsigned short* p = A + (size_t)row * K + kk;
  FragU u;
  u.q[0] = ((const uint4*)p)[0];          // K = kk .. kk+7
  u.q[1] = ((const uint4*)(p + 8))[0];    // K = kk+8 .. kk+15
  return u.v;
}

// A fragment with fp32 -> bf16 hardware conversion on load.
__device__ __forceinline__ v16bf load_afrag_f32(const float* A, size_t rowstride,
                                                int m0, int k0, int lane) {
  int row = m0 + (lane & 15);
  int kk  = k0 + ((lane & 16) ? 16 : 0);
  const float* p = A + (size_t)row * rowstride + kk;
  FragU u;
#pragma unroll
  for (int j = 0; j < 8; ++j) {
    u.e[j]     = (__bf16)p[j];
    u.e[8 + j] = (__bf16)p[8 + j];
  }
  return u.v;
}

// B fragment (32x16, bf16) from transposed weight stored [N][K] row-major.
__device__ __forceinline__ v16bf load_bfragT(const unsigned short* WT, int n0,
                                             int K, int k0, int lane) {
  const unsigned short* p =
      WT + (size_t)(n0 + (lane & 15)) * K + k0 + ((lane & 16) ? 16 : 0);
  FragU u;
  u.q[0] = ((const uint4*)p)[0];
  u.q[1] = ((const uint4*)p)[1];
  return u.v;
}

// ---------------------------------------------------------------------------
// Problem sizes
// ---------------------------------------------------------------------------
#define BB      64
#define TT      1024
#define IN_DIM  64
#define DD      512
#define HH      1536      /* 3*D */
#define OUT_DIM 64
#define TOKT    1040      /* padded token rows per batch (1025 used) */
#define NWG     32        /* scan workgroups (16 cols each) */

#if defined(__has_builtin)
#if __has_builtin(__builtin_amdgcn_global_load_async_to_lds_b128)
#define ASYNC_XW 1
#endif
#endif
#ifndef ASYNC_XW
#define ASYNC_XW 0
#endif

// ---------------------------------------------------------------------------
// Prep kernels
// ---------------------------------------------------------------------------
__global__ void k_transpose(const float* __restrict__ src,
                            unsigned short* __restrict__ dst, int R, int C) {
  int i = blockIdx.x * 256 + threadIdx.x;
  if (i >= R * C) return;
  int r = i / C, c = i - r * C;
  dst[(size_t)c * R + r] = f2bf(src[i]);
}

__global__ void k_init(float* __restrict__ hf32, unsigned short* __restrict__ hbf,
                       unsigned* __restrict__ counter,
                       unsigned short* __restrict__ tok,
                       const float* __restrict__ sink) {
  int i = blockIdx.x * 256 + threadIdx.x;
  if (i == 0) *counter = 0u;
  if (i < 2 * BB * DD) { hf32[i] = 0.0f; hbf[i] = 0; }
  if (i < BB * DD) {                       // sink token at t=0 for every b
    int b = i >> 9, d = i & (DD - 1);
    tok[((size_t)b * TOKT + 0) * DD + d] = f2bf(sink[d]);
  }
}

// ---------------------------------------------------------------------------
// Embed: tok[b][1+t][:] = x[b][t][:] @ W_embed + b_embed   (bf16 out)
// ---------------------------------------------------------------------------
__global__ void __launch_bounds__(128)
k_embed(const float* __restrict__ x, const unsigned short* __restrict__ WembT,
        const float* __restrict__ b_embed, unsigned short* __restrict__ tok) {
  int wid  = blockIdx.x * 4 + (threadIdx.x >> 5);
  int lane = threadIdx.x & 31;
  int mtile = wid >> 5;                 // 4096 m-tiles (B*T/16)
  int ntile = wid & 31;                 // 32 n-tiles (D/16)
  int m0 = mtile * 16, n0 = ntile * 16;
  v8f acc = {};
#pragma unroll
  for (int k0 = 0; k0 < IN_DIM; k0 += 32) {
    v16bf a = load_afrag_f32(x, IN_DIM, m0, k0, lane);
    v16bf b = load_bfragT(WembT, n0, IN_DIM, k0, lane);
    acc = wmma_bf16(a, b, acc);
  }
  int col = n0 + (lane & 15);
  float bias = b_embed[col];
#pragma unroll
  for (int r = 0; r < 8; ++r) {
    int m = m0 + r + ((lane & 16) ? 8 : 0);
    int b = m >> 10, t = m & 1023;
    tok[((size_t)b * TOKT + 1 + t) * DD + col] = f2bf(acc[r] + bias);
  }
}

// ---------------------------------------------------------------------------
// xw[t][b][h] = tok[b][t][:] @ Wi + bi   (bf16, t-major for the scan)
// ---------------------------------------------------------------------------
__global__ void __launch_bounds__(128)
k_xw(const unsigned short* __restrict__ tok, const unsigned short* __restrict__ WiT,
     const float* __restrict__ bi, unsigned short* __restrict__ xw) {
  int wid  = blockIdx.x * 4 + (threadIdx.x >> 5);
  int lane = threadIdx.x & 31;
  int htile = wid % 96;                  // H/16
  int rest  = wid / 96;
  int ttile = rest % 65;                 // ceil(1025/16)
  int b     = rest / 65;
  int t0 = ttile * 16, h0 = htile * 16;
  const unsigned short* A = tok + (size_t)b * TOKT * DD;
  v8f acc = {};
  for (int k0 = 0; k0 < DD; k0 += 32) {
    v16bf a  = load_afrag(A, t0, DD, k0, lane);
    v16bf bm = load_bfragT(WiT, h0, DD, k0, lane);
    acc = wmma_bf16(a, bm, acc);
  }
  int h = h0 + (lane & 15);
  float bias = bi[h];
#pragma unroll
  for (int r = 0; r < 8; ++r) {
    int t = t0 + r + ((lane & 16) ? 8 : 0);
    if (t < TT + 1)
      xw[((size_t)t * BB + b) * HH + h] = f2bf(acc[r] + bias);
  }
}

// ---------------------------------------------------------------------------
// Persistent recurrent scan: 32 WGs, each owns 16 columns of h.
//   Per step: hw = h @ Wh (bf16 WMMA, WhT stripe staged in LDS once),
//   gate inputs double-buffered into LDS via async global->LDS (ASYNCcnt),
//   gates in fp32 VALU, h ping-pong through global, one device barrier/step.
// ---------------------------------------------------------------------------
__global__ void __launch_bounds__(128, 1)
k_scan(const unsigned short* __restrict__ xw, const unsigned short* __restrict__ WhT,
       unsigned short* __restrict__ hbf, float* __restrict__ hf32,
       unsigned short* __restrict__ hs, unsigned* __restrict__ counter) {
  __shared__ unsigned short ldsWh[48][520];   // 3 gates x 16 cols, K=512 (+16B pad)
#if ASYNC_XW
  __shared__ unsigned short ldsXw[2][64][48]; // double-buffered gate inputs (12KB)
#endif

  int c0   = blockIdx.x * 16;
  int lane = threadIdx.x & 31;
  int w    = threadIdx.x >> 5;
  int m0   = w * 16;                          // each wave owns 16 batch rows

  // Stage this WG's constant WhT stripe into LDS (once for all 1025 steps).
  for (int idx = threadIdx.x; idx < 48 * 32; idx += 128) {
    int row = idx >> 5, chunk = idx & 31;     // 32 x 16B chunks = first 256 elems per row
    int g = row >> 4, j = row & 15;
    const uint4* src = (const uint4*)(WhT + (size_t)(g * DD + c0 + j) * DD) + chunk;
    *(uint4*)&ldsWh[row][chunk * 8] = *src;
  }
  __syncthreads();

#if ASYNC_XW
  // Issue async global->LDS copy of step t's 6KB xw stripe (3 b128 ops/thread).
  auto issue_xw = [&](int t, int buf) {
#pragma unroll
    for (int j = 0; j < 3; ++j) {
      int c = (int)threadIdx.x + j * 128;     // 384 x 16B chunks
      int b = c / 6, piece = c - b * 6;       // 6 chunks per batch row
      int g = piece >> 1, half = piece & 1;
      const unsigned short* gsrc =
          xw + ((size_t)t * BB + b) * HH + g * DD + c0 + half * 8;
      __builtin_amdgcn_global_load_async_to_lds_b128(
          (v4i*)gsrc, (v4i*)&ldsXw[buf][b][piece * 8], 0, 0);
    }
  };
  issue_xw(0, 0);
#endif

  unsigned target = 0;
  for (int t = 0; t <= TT; ++t) {
    int cur = t & 1, nxt = cur ^ 1;
    const unsigned short* h_in = hbf + (size_t)cur * BB * DD;

#if ASYNC_XW
    if (t < TT) issue_xw(t + 1, nxt);         // overlap with this step's GEMM
#else
    if (t < TT) {
      for (int i = threadIdx.x; i < 192; i += 128) {
        int row = i & 63, g = i >> 6;
        __builtin_prefetch(xw + ((size_t)(t + 1) * BB + row) * HH + g * DD + c0, 0, 0);
      }
    }
#endif

    // hw tiles for gates r,z,n : three 16x16 f32 accumulators per wave.
    v8f accR = {}, accZ = {}, accN = {};
    for (int k0 = 0; k0 < DD; k0 += 32) {
      v16bf a = load_afrag(h_in, m0, DD, k0, lane);
      int koff = k0 + ((lane & 16) ? 16 : 0);
      FragU ur, uz, un;
      {
        const unsigned short* p = &ldsWh[0  + (lane & 15)][koff];
        ur.q[0] = ((const uint4*)p)[0]; ur.q[1] = ((const uint4*)p)[1];
      }
      {
        const unsigned short* p = &ldsWh[16 + (lane & 15)][koff];
        uz.q[0] = ((const uint4*)p)[0]; uz.q[1] = ((const uint4*)p)[1];
      }
      {
        const unsigned short* p = &ldsWh[32 + (lane & 15)][koff];
        un.q[0] = ((const uint4*)p)[0]; un.q[1] = ((const uint4*)p)[1];
      }
      accR = wmma_bf16(a, ur.v, accR);
      accZ = wmma_bf16(a, uz.v, accZ);
      accN = wmma_bf16(a, un.v, accN);
    }

#if ASYNC_XW
    // Retire this step's async batch (next step's 3 ops may stay in flight),
    // then sync so every wave sees the whole WG's stripe.
    if (t < TT) asm volatile("s_wait_asynccnt 0x3" ::: "memory");
    else        asm volatile("s_wait_asynccnt 0x0" ::: "memory");
    __syncthreads();
#endif

    // Gates + state update (fp32), write new h stripe + history.
    int col = c0 + (lane & 15);
    int n   = lane & 15;
    const unsigned short* xwt = xw + (size_t)t * BB * HH;
    (void)xwt; (void)n;
#pragma unroll
    for (int r = 0; r < 8; ++r) {
      int b = m0 + r + ((lane & 16) ? 8 : 0);
#if ASYNC_XW
      float xr = bf2f(ldsXw[cur][b][n]);
      float xz = bf2f(ldsXw[cur][b][16 + n]);
      float xn = bf2f(ldsXw[cur][b][32 + n]);
#else
      float xr = bf2f(xwt[(size_t)b * HH + col]);
      float xz = bf2f(xwt[(size_t)b * HH + DD + col]);
      float xn = bf2f(xwt[(size_t)b * HH + 2 * DD + col]);
#endif
      float rr = sigmoidf(xr + accR[r]);
      float zz = sigmoidf(xz + accZ[r]);
      float nn = tanhf(xn + rr * accN[r]);
      float hold = hf32[(size_t)cur * BB * DD + (size_t)b * DD + col];
      float hnew = (1.0f - zz) * nn + zz * hold;
      hf32[(size_t)nxt * BB * DD + (size_t)b * DD + col] = hnew;
      unsigned short hb = f2bf(hnew);
      hbf[(size_t)nxt * BB * DD + (size_t)b * DD + col] = hb;
      if (t >= 1)
        hs[((size_t)b * TT + (t - 1)) * DD + col] = hb;   // history for unembed
    }

    // Device-wide barrier (release h writes, acquire everyone's).
    __threadfence();
    __syncthreads();
    target += NWG;
    if (threadIdx.x == 0) {
      atomicAdd(counter, 1u);
      while (__atomic_load_n(counter, __ATOMIC_ACQUIRE) < target)
        __builtin_amdgcn_s_sleep(1);
    }
    __syncthreads();
    __threadfence();
  }
}

// ---------------------------------------------------------------------------
// Unembed: out[b][t][:] = hs[b][t][:] @ W_unembed + b_unembed  (fp32 out)
// ---------------------------------------------------------------------------
__global__ void __launch_bounds__(128)
k_unembed(const unsigned short* __restrict__ hs, const unsigned short* __restrict__ WunT,
          const float* __restrict__ b_unembed, float* __restrict__ out) {
  int wid  = blockIdx.x * 4 + (threadIdx.x >> 5);
  int lane = threadIdx.x & 31;
  int mtile = wid >> 2;                 // 4096 m-tiles (B*T/16)
  int ntile = wid & 3;                  // 4 n-tiles  (OUT/16)
  int m0 = mtile * 16, n0 = ntile * 16;
  v8f acc = {};
  for (int k0 = 0; k0 < DD; k0 += 32) {
    v16bf a = load_afrag(hs, m0, DD, k0, lane);
    v16bf b = load_bfragT(WunT, n0, DD, k0, lane);
    acc = wmma_bf16(a, b, acc);
  }
  int col = n0 + (lane & 15);
  float bias = b_unembed[col];
#pragma unroll
  for (int r = 0; r < 8; ++r) {
    int m = m0 + r + ((lane & 16) ? 8 : 0);
    out[(size_t)m * OUT_DIM + col] = acc[r] + bias;
  }
}

// ---------------------------------------------------------------------------
// Host launch
// ---------------------------------------------------------------------------
extern "C" void kernel_launch(void* const* d_in, const int* in_sizes, int n_in,
                              void* d_out, int out_size, void* d_ws, size_t ws_size,
                              hipStream_t stream) {
  (void)in_sizes; (void)n_in; (void)out_size; (void)ws_size;

  const float* x         = (const float*)d_in[0];
  const float* W_embed   = (const float*)d_in[1];
  const float* b_embed   = (const float*)d_in[2];
  const float* sink      = (const float*)d_in[3];
  const float* Wi        = (const float*)d_in[4];
  const float* bi        = (const float*)d_in[5];
  const float* Wh        = (const float*)d_in[6];
  const float* W_unembed = (const float*)d_in[7];
  const float* b_unembed = (const float*)d_in[8];
  float* out = (float*)d_out;

  char* ws = (char*)d_ws;
  size_t off = 0;
  auto alloc = [&](size_t bytes) -> void* {
    void* p = ws + off;
    off = (off + bytes + 255) & ~(size_t)255;
    return p;
  };
  unsigned short* WiT   = (unsigned short*)alloc((size_t)DD * HH * 2);
  unsigned short* WhT   = (unsigned short*)alloc((size_t)DD * HH * 2);
  unsigned short* WembT = (unsigned short*)alloc((size_t)IN_DIM * DD * 2);
  unsigned short* WunT  = (unsigned short*)alloc((size_t)DD * OUT_DIM * 2);
  unsigned short* hbf   = (unsigned short*)alloc((size_t)2 * BB * DD * 2);
  float*          hf32  = (float*)alloc((size_t)2 * BB * DD * 4);
  unsigned*       ctr   = (unsigned*)alloc(256);
  unsigned short* xw    = (unsigned short*)alloc((size_t)(TT + 1) * BB * HH * 2);
  unsigned short* tok   = (unsigned short*)alloc((size_t)BB * TOKT * DD * 2);
  unsigned short* hs    = tok;   // tok dead after k_xw; reuse as h history

  // Weight transposes (fp32 -> bf16, [N][K] layout for B-fragments).
  k_transpose<<<(DD * HH + 255) / 256, 256, 0, stream>>>(Wi, WiT, DD, HH);
  k_transpose<<<(DD * HH + 255) / 256, 256, 0, stream>>>(Wh, WhT, DD, HH);
  k_transpose<<<(IN_DIM * DD + 255) / 256, 256, 0, stream>>>(W_embed, WembT, IN_DIM, DD);
  k_transpose<<<(DD * OUT_DIM + 255) / 256, 256, 0, stream>>>(W_unembed, WunT, DD, OUT_DIM);

  // Zero h / counter, place sink token (re-done every launch: graph-safe).
  k_init<<<(2 * BB * DD + 255) / 256, 256, 0, stream>>>(hf32, hbf, ctr, tok, sink);

  // Parallel phases.
  k_embed<<<(4096 * 32) / 4, 128, 0, stream>>>(x, WembT, b_embed, tok);
  k_xw<<<(64 * 65 * 96) / 4, 128, 0, stream>>>(tok, WiT, bi, xw);

  // Serial recurrence: 32 persistent WGs, 1025 steps.
  k_scan<<<NWG, 128, 0, stream>>>(xw, WhT, hbf, hf32, hs, ctr);

  // Output projection.
  k_unembed<<<(4096 * 4) / 4, 128, 0, stream>>>(hs, WunT, b_unembed, out);
}
